// OutsideEncoder_3865470566621
// MI455X (gfx1250) — compile-verified, matrix operand order checked by reading, not run on hardware
//
#include <hip/hip_runtime.h>
#include <hip/hip_bf16.h>
#include <math.h>

// ---------------------------------------------------------------------------
// Types for CDNA5 WMMA (wave32)
// ---------------------------------------------------------------------------
typedef __bf16 bf16_t;
typedef __attribute__((ext_vector_type(16))) __bf16 v16bf;
typedef __attribute__((ext_vector_type(8)))  float  v8f;

// ---------------------------------------------------------------------------
// bf16 helpers (round-to-nearest-even)
// ---------------------------------------------------------------------------
__device__ __forceinline__ unsigned short bfbits(float f) {
  unsigned u = __builtin_bit_cast(unsigned, f);
  return (unsigned short)((u + 0x7FFFu + ((u >> 16) & 1u)) >> 16);
}
__device__ __forceinline__ bf16_t f2bf(float f) {
  unsigned short s = bfbits(f);
  return __builtin_bit_cast(bf16_t, s);
}
__device__ __forceinline__ float bf2f(bf16_t h) {
  unsigned u = ((unsigned)__builtin_bit_cast(unsigned short, h)) << 16;
  return __builtin_bit_cast(float, u);
}
// Pack two f32 -> two bf16 (RNE) in one dword using v_perm_b32.
__device__ __forceinline__ unsigned pack_bf16(float x, float y) {
  unsigned ux = __builtin_bit_cast(unsigned, x);
  unsigned uy = __builtin_bit_cast(unsigned, y);
  ux += 0x7FFFu + ((ux >> 16) & 1u);
  uy += 0x7FFFu + ((uy >> 16) & 1u);
  // dst bytes: [1:0] = ux[3:2], [3:2] = uy[3:2]  (low half = bf16(x))
  return __builtin_amdgcn_perm(uy, ux, 0x07060302);
}

// ---------------------------------------------------------------------------
// f32 -> bf16 bulk convert (weights)
// ---------------------------------------------------------------------------
__global__ void __launch_bounds__(256) cvt_bf16_kernel(const float* __restrict__ in,
                                                       bf16_t* __restrict__ out, long n) {
  long i = (long)blockIdx.x * 256 + threadIdx.x;
  if (i < n) out[i] = f2bf(in[i]);
}

// ---------------------------------------------------------------------------
// Build x (32768 x 768): row r = n*2+q ; n=(b*32+c)*2+k
//   q==0 -> parent_ij[b] ; q==1 -> child_ikj[b,c,1-k]
// ---------------------------------------------------------------------------
__global__ void __launch_bounds__(256) build_x_kernel(const float* __restrict__ parent,
                                                      const float* __restrict__ child,
                                                      float* __restrict__ xf) {
  long idx = (long)blockIdx.x * 256 + threadIdx.x;  // < 32768*768
  int d = (int)(idx % 768);
  long r = idx / 768;
  long n = r >> 1;
  int q = (int)(r & 1);
  long b = n >> 6;                 // C*2 = 64
  int rem = (int)(n & 63);
  int c = rem >> 1, k = rem & 1;
  float v;
  if (q == 0) v = parent[b * 768 + d];
  else        v = child[(((b * 32 + c) * 2 + (1 - k)) * 768) + d];
  xf[idx] = v;
}

// qk_in = x + pos_emb[l][role];  role = (q==0) ? 0 : 1+k
__global__ void __launch_bounds__(256) build_qkin_kernel(const float* __restrict__ xf,
                                                         const float* __restrict__ pe,
                                                         float* __restrict__ qkin) {
  long idx = (long)blockIdx.x * 256 + threadIdx.x;
  int d = (int)(idx % 768);
  long r = idx / 768;
  long n = r >> 1;
  int q = (int)(r & 1);
  int k = (int)(n & 1);
  int role = (q == 0) ? 0 : (1 + k);
  qkin[idx] = xf[idx] + pe[role * 768 + d];
}

// ---------------------------------------------------------------------------
// Tiled WMMA GEMM:  C[M,N] = act( A[M,K] * W[N,K]^T + bias[N] )
//   A: float or bf16 (row-major, lda elems)  -> staged through regs w/ convert
//   W: bf16 (row-major N x K)                -> staged via async global->LDS
//   C: float or bf16 (ldc elems)
// Block: 256 threads (8 wave32 waves). Tile 128x128, BK=32, double-buffered.
// Wave grid 4(M) x 2(N): each wave owns 32 rows x 64 cols = 2x4 WMMA frags.
// ---------------------------------------------------------------------------
constexpr int BM = 128, BN = 128, BK = 32;
constexpr int LSTR = 40;  // LDS row stride in bf16 elems (80B) -> conflict-free

template <typename T> struct AVec;
template <> struct AVec<float>  { float4 a, b; };
template <> struct AVec<bf16_t> { uint4 q; };

__device__ __forceinline__ void ldAvec(AVec<float>& d, const float* p) {
  const float4* s = (const float4*)p;
  d.a = s[0]; d.b = s[1];
}
__device__ __forceinline__ void ldAvec(AVec<bf16_t>& d, const bf16_t* p) {
  d.q = *(const uint4*)p;
}
__device__ __forceinline__ void stAvec(bf16_t* dst, const AVec<float>& d) {
  uint4 o;
  o.x = pack_bf16(d.a.x, d.a.y);
  o.y = pack_bf16(d.a.z, d.a.w);
  o.z = pack_bf16(d.b.x, d.b.y);
  o.w = pack_bf16(d.b.z, d.b.w);
  *(uint4*)dst = o;
}
__device__ __forceinline__ void stAvec(bf16_t* dst, const AVec<bf16_t>& d) {
  *(uint4*)dst = d.q;
}

// Async global->LDS 16B copy (gfx1250): vdst = LDS byte offset, vaddr = 64b addr.
__device__ __forceinline__ void async_ld_b128(const bf16_t* lds_ptr, const bf16_t* gptr) {
  unsigned lds_off = (unsigned)(uintptr_t)lds_ptr;  // LDS aperture: addr[31:0] = LDS offset
  asm volatile("global_load_async_to_lds_b128 %0, %1, off"
               :: "v"(lds_off), "v"((unsigned long long)(uintptr_t)gptr)
               : "memory");
}
__device__ __forceinline__ void wait_async0() {
  asm volatile("s_wait_asynccnt 0" ::: "memory");
}

union FragU { uint4 q[2]; v16bf v; };

// 16-bit A/B fragment layout: lanes 0-15 -> row (lane&15), K = 0..7 & 16..23;
// lanes 16-31 -> same rows, K = 8..15 & 24..31 (per ISA 7.12.2).
__device__ __forceinline__ v16bf load_frag(const bf16_t* rowbase, int lane) {
  const bf16_t* p = rowbase + (lane & 15) * LSTR + (lane >> 4) * 8;
  FragU f;
  f.q[0] = *(const uint4*)p;         // K = khalf .. khalf+7
  f.q[1] = *(const uint4*)(p + 16);  // K = khalf+16 .. khalf+23
  return f.v;
}

__device__ __forceinline__ void storeC(float*  p, float v) { *p = v; }
__device__ __forceinline__ void storeC(bf16_t* p, float v) { *p = f2bf(v); }

template <typename AT, typename CT, bool GELU>
__global__ void __launch_bounds__(256) gemm_wmma(const AT* __restrict__ A, int lda,
                                                 const bf16_t* __restrict__ W,
                                                 const float* __restrict__ bias,
                                                 CT* __restrict__ C, int ldc, int K) {
  __shared__ bf16_t sA[2][BM * LSTR];
  __shared__ bf16_t sB[2][BN * LSTR];
  const int tid = threadIdx.x;
  const int wave = tid >> 5, lane = tid & 31;
  const long bm = (long)blockIdx.x * BM;
  const long bn = (long)blockIdx.y * BN;
  const int wm = (wave >> 1) * 32;  // 4 waves along M
  const int wn = (wave & 1) * 64;   // 2 waves along N

  // Per-thread staging coordinates: vector v = i*256+tid (512 x 16B per tile),
  // row = v>>2 (4 vectors per 32-elem row), col = (v&3)*8.
  int srow[2], scol[2];
#pragma unroll
  for (int i = 0; i < 2; ++i) {
    int v = i * 256 + tid;
    srow[i] = v >> 2;
    scol[i] = (v & 3) << 3;
  }

  v8f acc[2][4] = {};
  AVec<AT> ar[2];

  auto loadA = [&](int k0) {
#pragma unroll
    for (int i = 0; i < 2; ++i)
      ldAvec(ar[i], A + (bm + srow[i]) * (long)lda + k0 + scol[i]);
  };
  auto storeA = [&](int buf) {
#pragma unroll
    for (int i = 0; i < 2; ++i)
      stAvec(&sA[buf][srow[i] * LSTR + scol[i]], ar[i]);
  };
  auto asyncW = [&](int k0, int buf) {
#pragma unroll
    for (int i = 0; i < 2; ++i)
      async_ld_b128(&sB[buf][srow[i] * LSTR + scol[i]],
                    W + (bn + srow[i]) * (long)K + k0 + scol[i]);
  };
  auto compute = [&](int buf) {
    v16bf af[2], bfr[4];
#pragma unroll
    for (int i = 0; i < 2; ++i) af[i] = load_frag(&sA[buf][(wm + i * 16) * LSTR], lane);
#pragma unroll
    for (int j = 0; j < 4; ++j) bfr[j] = load_frag(&sB[buf][(wn + j * 16) * LSTR], lane);
#pragma unroll
    for (int i = 0; i < 2; ++i)
#pragma unroll
      for (int j = 0; j < 4; ++j)
        acc[i][j] = __builtin_amdgcn_wmma_f32_16x16x32_bf16(
            false, af[i], false, bfr[j], (short)0, acc[i][j], false, false);
  };

  // ---- software pipeline (2-deep) ----
  loadA(0);
  asyncW(0, 0);
  storeA(0);
  wait_async0();
  __syncthreads();  // tile 0 ready

  const int nt = K / BK;
  for (int t = 1; t < nt; ++t) {
    const int cur = (t - 1) & 1, nxt = t & 1;
    const int k0 = t * BK;
    loadA(k0);        // global loads in flight while we compute
    asyncW(k0, nxt);  // async global->LDS for W tile t
    if (t + 1 < nt)   // prefetch A one more tile ahead -> global_prefetch_b8
      __builtin_prefetch((const void*)(A + (bm + srow[0]) * (long)lda + k0 + BK + scol[0]), 0, 1);
    compute(cur);
    storeA(nxt);
    wait_async0();
    __syncthreads();  // tile t ready; tile t-1 fully consumed
  }
  compute((nt - 1) & 1);

  // Epilogue: C/D layout -> lane: N=lane&15, M = vreg + 8*(lane>>4)
#pragma unroll
  for (int j = 0; j < 4; ++j) {
    long col = bn + wn + j * 16 + (lane & 15);
    float bv = bias[col];
#pragma unroll
    for (int i = 0; i < 2; ++i) {
      long row0 = bm + wm + i * 16 + (lane >> 4) * 8;
#pragma unroll
      for (int v = 0; v < 8; ++v) {
        float val = acc[i][j][v] + bv;
        if (GELU) val = 0.5f * val * (1.0f + erff(val * 0.70710678118654752f));
        storeC(C + (row0 + v) * (long)ldc + col, val);
      }
    }
  }
}

// ---------------------------------------------------------------------------
// Attention over length-2 sequences. qkv: [32768][2304] bf16 (q|k|v blocks).
// One wave per token-pair n; lanes cover 64 head dims (2 per lane).
// ---------------------------------------------------------------------------
__global__ void __launch_bounds__(256) attn_kernel(const bf16_t* __restrict__ qkv,
                                                   float* __restrict__ o) {
  const int wave = threadIdx.x >> 5, lane = threadIdx.x & 31;
  const long n = (long)blockIdx.x * 8 + wave;  // < 16384
  const bf16_t* r0 = qkv + n * 2 * 2304;
  const bf16_t* r1 = r0 + 2304;
  float* o0 = o + n * 2 * 768;
  float* o1 = o0 + 768;
  for (int h = 0; h < 12; ++h) {
    const int d = h * 64 + lane * 2;
    float q0a = bf2f(r0[d]),        q0b = bf2f(r0[d + 1]);
    float q1a = bf2f(r1[d]),        q1b = bf2f(r1[d + 1]);
    float k0a = bf2f(r0[768 + d]),  k0b = bf2f(r0[768 + d + 1]);
    float k1a = bf2f(r1[768 + d]),  k1b = bf2f(r1[768 + d + 1]);
    float v0a = bf2f(r0[1536 + d]), v0b = bf2f(r0[1536 + d + 1]);
    float v1a = bf2f(r1[1536 + d]), v1b = bf2f(r1[1536 + d + 1]);
    float s00 = q0a * k0a + q0b * k0b;
    float s01 = q0a * k1a + q0b * k1b;
    float s10 = q1a * k0a + q1b * k0b;
    float s11 = q1a * k1a + q1b * k1b;
#pragma unroll
    for (int off = 16; off > 0; off >>= 1) {
      s00 += __shfl_xor(s00, off);
      s01 += __shfl_xor(s01, off);
      s10 += __shfl_xor(s10, off);
      s11 += __shfl_xor(s11, off);
    }
    const float sc = 0.125f;  // 1/sqrt(64)
    s00 *= sc; s01 *= sc; s10 *= sc; s11 *= sc;
    float m0 = fmaxf(s00, s01), m1 = fmaxf(s10, s11);
    float e00 = __expf(s00 - m0), e01 = __expf(s01 - m0);
    float e10 = __expf(s10 - m1), e11 = __expf(s11 - m1);
    float i0 = 1.f / (e00 + e01), i1 = 1.f / (e10 + e11);
    float a00 = e00 * i0, a01 = e01 * i0, a10 = e10 * i1, a11 = e11 * i1;
    o0[d]     = a00 * v0a + a01 * v1a;
    o0[d + 1] = a00 * v0b + a01 * v1b;
    o1[d]     = a10 * v0a + a11 * v1a;
    o1[d + 1] = a10 * v0b + a11 * v1b;
  }
}

// ---------------------------------------------------------------------------
// out = inorm(a + b) per 768-wide row. One 256-thread block per row.
// ---------------------------------------------------------------------------
__global__ void __launch_bounds__(256) add_inorm_kernel(const float* __restrict__ a, int lda,
                                                        const float* __restrict__ b, int ldb,
                                                        float* __restrict__ out, int ldo) {
  const long row = blockIdx.x;
  const float* pa = a + row * (long)lda;
  const float* pb = b + row * (long)ldb;
  float* po = out + row * (long)ldo;
  const int tid = threadIdx.x, wave = tid >> 5, lane = tid & 31;
  float loc[3];
  float s = 0.f, ss = 0.f;
#pragma unroll
  for (int i = 0; i < 3; ++i) {
    int idx = tid + i * 256;
    float v = pa[idx] + pb[idx];
    loc[i] = v; s += v; ss += v * v;
  }
#pragma unroll
  for (int off = 16; off > 0; off >>= 1) { s += __shfl_xor(s, off); ss += __shfl_xor(ss, off); }
  __shared__ float wsum[8], wsq[8];
  if (lane == 0) { wsum[wave] = s; wsq[wave] = ss; }
  __syncthreads();
  s = 0.f; ss = 0.f;
#pragma unroll
  for (int w = 0; w < 8; ++w) { s += wsum[w]; ss += wsq[w]; }
  const float mean = s * (1.f / 768.f);
  const float var = ss * (1.f / 768.f) - mean * mean;
  const float rstd = rsqrtf(var + 1e-5f);
#pragma unroll
  for (int i = 0; i < 3; ++i) {
    int idx = tid + i * 256;
    po[idx] = (loc[i] - mean) * rstd;
  }
}

// ---------------------------------------------------------------------------
// outside_scores: one wave per (b,c); two 512-dim dots.
// ---------------------------------------------------------------------------
__global__ void __launch_bounds__(256) score_dot_kernel(const float* __restrict__ pc,
                                                        const float* __restrict__ rc,
                                                        const float* __restrict__ lc,
                                                        float* __restrict__ outs) {
  const int wave = threadIdx.x >> 5, lane = threadIdx.x & 31;
  const long bc = (long)blockIdx.x * 8 + wave;  // < 8192
  const long b = bc >> 5;                       // C = 32
  const float* p = pc + b * 512;
  const float* r = rc + bc * 512;
  const float* l = lc + bc * 512;
  float sl = 0.f, sr = 0.f;
#pragma unroll 4
  for (int i = lane; i < 512; i += 32) {
    float pv = p[i];
    sl += pv * r[i];
    sr += pv * l[i];
  }
#pragma unroll
  for (int off = 16; off > 0; off >>= 1) { sl += __shfl_xor(sl, off); sr += __shfl_xor(sr, off); }
  if (lane == 0) {
    const float scale = 0.044194173824159216f;  // 1/sqrt(512)
    outs[bc * 2 + 0] = sl * scale;
    outs[bc * 2 + 1] = sr * scale;
  }
}

// ---------------------------------------------------------------------------
// Host-side orchestration
// ---------------------------------------------------------------------------
extern "C" void kernel_launch(void* const* d_in, const int* in_sizes, int n_in,
                              void* d_out, int out_size, void* d_ws, size_t ws_size,
                              hipStream_t stream) {
  (void)in_sizes; (void)n_in; (void)out_size; (void)ws_size;
  const float* parent_ij  = (const float*)d_in[0];
  const float* child_ikj  = (const float*)d_in[1];
  // d_in[2], d_in[3] (parent_scores, child_scores) unused by the reference
  const float* attn_in_w  = (const float*)d_in[4];
  const float* attn_in_b  = (const float*)d_in[5];
  const float* attn_out_w = (const float*)d_in[6];
  const float* attn_out_b = (const float*)d_in[7];
  const float* lin1_w     = (const float*)d_in[8];
  const float* lin1_b     = (const float*)d_in[9];
  const float* lin2_w     = (const float*)d_in[10];
  const float* lin2_b     = (const float*)d_in[11];
  const float* pos_emb    = (const float*)d_in[12];
  const float* parent_w1  = (const float*)d_in[13];
  const float* parent_b1  = (const float*)d_in[14];
  const float* parent_w2  = (const float*)d_in[15];
  const float* parent_b2  = (const float*)d_in[16];
  const float* left_w1    = (const float*)d_in[17];
  const float* left_b1    = (const float*)d_in[18];
  const float* left_w2    = (const float*)d_in[19];
  const float* left_b2    = (const float*)d_in[20];
  const float* right_w1   = (const float*)d_in[21];
  const float* right_b1   = (const float*)d_in[22];
  const float* right_w2   = (const float*)d_in[23];
  const float* right_b2   = (const float*)d_in[24];

  float* out_scores = (float*)d_out;           // (B,C,2) = 16384
  float* out_e      = (float*)d_out + 16384;   // (B,C,2,D)

  char* wsp = (char*)d_ws;
  size_t off = 0;
  auto alloc = [&](size_t bytes) -> void* {
    off = (off + 255) & ~(size_t)255;
    void* p = wsp + off;
    off += bytes;
    return p;
  };
  float* xf    = (float*)alloc((size_t)32768 * 768 * 4);   // x (f32)
  float* regA  = (float*)alloc((size_t)32768 * 768 * 4);   // qk_in f32 / attn-out f32
  char*  regB  = (char*) alloc((size_t)32768 * 3072 * 2);  // scoring scratch / qkv bf16 / h bf16
  float* regC  = (float*)alloc((size_t)32768 * 768 * 4);   // proj / mlp2 out f32
  bf16_t* w_ain  = (bf16_t*)alloc((size_t)3 * 2304 * 768 * 2);
  bf16_t* w_aout = (bf16_t*)alloc((size_t)3 * 768 * 768 * 2);
  bf16_t* w_l1   = (bf16_t*)alloc((size_t)3 * 3072 * 768 * 2);
  bf16_t* w_l2   = (bf16_t*)alloc((size_t)3 * 768 * 3072 * 2);
  bf16_t* w_pw1  = (bf16_t*)alloc((size_t)768 * 768 * 2);
  bf16_t* w_pw2  = (bf16_t*)alloc((size_t)512 * 768 * 2);
  bf16_t* w_lw1  = (bf16_t*)alloc((size_t)768 * 768 * 2);
  bf16_t* w_lw2  = (bf16_t*)alloc((size_t)512 * 768 * 2);
  bf16_t* w_rw1  = (bf16_t*)alloc((size_t)768 * 768 * 2);
  bf16_t* w_rw2  = (bf16_t*)alloc((size_t)512 * 768 * 2);

  auto cvt = [&](const float* src, bf16_t* dst, long n) {
    cvt_bf16_kernel<<<dim3((unsigned)((n + 255) / 256)), dim3(256), 0, stream>>>(src, dst, n);
  };
  cvt(attn_in_w,  w_ain,  3L * 2304 * 768);
  cvt(attn_out_w, w_aout, 3L * 768 * 768);
  cvt(lin1_w,     w_l1,   3L * 3072 * 768);
  cvt(lin2_w,     w_l2,   3L * 768 * 3072);
  cvt(parent_w1,  w_pw1,  768L * 768);
  cvt(parent_w2,  w_pw2,  512L * 768);
  cvt(left_w1,    w_lw1,  768L * 768);
  cvt(left_w2,    w_lw2,  512L * 768);
  cvt(right_w1,   w_rw1,  768L * 768);
  cvt(right_w2,   w_rw2,  512L * 768);

  build_x_kernel<<<dim3(32768 * 768 / 256), 256, 0, stream>>>(parent_ij, child_ikj, xf);

  // ----- scoring MLPs + dots (scratch lives in regB, before the layer loop) -----
  bf16_t* hid_p = (bf16_t*)regB;
  bf16_t* hid_r = hid_p + (size_t)256 * 768;
  bf16_t* hid_l = hid_r + (size_t)8192 * 768;
  float*  pcb   = (float*)(hid_l + (size_t)8192 * 768);
  float*  rcb   = pcb + (size_t)256 * 512;
  float*  lcb   = rcb + (size_t)8192 * 512;

  gemm_wmma<float, bf16_t, true><<<dim3(2, 6), 256, 0, stream>>>(
      parent_ij, 768, w_pw1, parent_b1, hid_p, 768, 768);
  gemm_wmma<bf16_t, float, false><<<dim3(2, 4), 256, 0, stream>>>(
      hid_p, 768, w_pw2, parent_b2, pcb, 512, 768);
  gemm_wmma<float, bf16_t, true><<<dim3(64, 6), 256, 0, stream>>>(
      child_ikj + 768, 1536, w_rw1, right_b1, hid_r, 768, 768);   // child[:,:,1,:]
  gemm_wmma<bf16_t, float, false><<<dim3(64, 4), 256, 0, stream>>>(
      hid_r, 768, w_rw2, right_b2, rcb, 512, 768);
  gemm_wmma<float, bf16_t, true><<<dim3(64, 6), 256, 0, stream>>>(
      child_ikj, 1536, w_lw1, left_b1, hid_l, 768, 768);          // child[:,:,0,:]
  gemm_wmma<bf16_t, float, false><<<dim3(64, 4), 256, 0, stream>>>(
      hid_l, 768, w_lw2, left_b2, lcb, 512, 768);
  score_dot_kernel<<<dim3(8192 / 8), 256, 0, stream>>>(pcb, rcb, lcb, out_scores);

  // ----- transformer layers -----
  bf16_t* qkv  = (bf16_t*)regB;  // [32768][2304]
  bf16_t* hbuf = (bf16_t*)regB;  // [32768][3072] (reuses region after attention)
  for (int l = 0; l < 3; ++l) {
    build_qkin_kernel<<<dim3(32768 * 768 / 256), 256, 0, stream>>>(
        xf, pos_emb + (size_t)l * 4 * 768, regA);
    // QK projection: (32768,768) x (1536,768)^T -> qkv[:, 0:1536]
    gemm_wmma<float, bf16_t, false><<<dim3(256, 12), 256, 0, stream>>>(
        regA, 768, w_ain + (size_t)l * 2304 * 768, attn_in_b + (size_t)l * 2304,
        qkv, 2304, 768);
    // V projection (from x, not qk_in): -> qkv[:, 1536:2304]
    gemm_wmma<float, bf16_t, false><<<dim3(256, 6), 256, 0, stream>>>(
        xf, 768, w_ain + (size_t)l * 2304 * 768 + (size_t)1536 * 768,
        attn_in_b + (size_t)l * 2304 + 1536, qkv + 1536, 2304, 768);
    attn_kernel<<<dim3(16384 / 8), 256, 0, stream>>>(qkv, regA);
    // Output projection
    gemm_wmma<float, float, false><<<dim3(256, 6), 256, 0, stream>>>(
        regA, 768, w_aout + (size_t)l * 768 * 768, attn_out_b + (size_t)l * 768,
        regC, 768, 768);
    add_inorm_kernel<<<dim3(32768), 256, 0, stream>>>(xf, 768, regC, 768, xf, 768);
    // MLP
    gemm_wmma<float, bf16_t, true><<<dim3(256, 24), 256, 0, stream>>>(
        xf, 768, w_l1 + (size_t)l * 3072 * 768, lin1_b + (size_t)l * 3072,
        hbuf, 3072, 768);
    gemm_wmma<bf16_t, float, false><<<dim3(256, 6), 256, 0, stream>>>(
        hbuf, 3072, w_l2 + (size_t)l * 768 * 3072, lin2_b + (size_t)l * 768,
        regC, 768, 3072);
    add_inorm_kernel<<<dim3(32768), 256, 0, stream>>>(xf, 768, regC, 768, xf, 768);
  }

  // out_e_ij = inorm(x[:,0,:] + x[:,1,:])
  add_inorm_kernel<<<dim3(16384), 256, 0, stream>>>(xf, 1536, xf + 768, 1536, out_e, 768);
}
